// CustomRNN_59270548684956
// MI455X (gfx1250) — matrix operand order repaired and checked by reading, not compiled, not run on hardware
//
#include <hip/hip_runtime.h>
#include <stdint.h>

#define B_ 128
#define T_ 64
#define D_ 256
#define H_ 256

typedef __attribute__((ext_vector_type(16))) __bf16 v16bf;
typedef __attribute__((ext_vector_type(8)))  float  v8f;
typedef __attribute__((ext_vector_type(4)))  int    v4i_t;

// explicit global-address-space views (force global_load/store, never FLAT)
typedef __attribute__((address_space(1))) const float  gcfloat_t;
typedef __attribute__((address_space(1))) const __bf16 gcbf_t;
typedef __attribute__((address_space(1))) float        gfloat_t;
typedef __attribute__((address_space(1))) __bf16       gbf_t;

#define BM 64
#define BN 32
#define BK 32
#define LDP 40   // LDS row pitch in bf16 elems (80B: 16B-aligned, conflict-friendly)

// gfx1250 async global->LDS copy (ASYNCcnt path), with safe fallback
#if defined(__AMDGCN__) && __has_builtin(__builtin_amdgcn_global_load_async_to_lds_b128)
#define HAVE_ASYNC_LDS 1
#else
#define HAVE_ASYNC_LDS 0
#endif

static __device__ __forceinline__ void async_copy_b128(const void* g, void* l) {
#if HAVE_ASYNC_LDS
  __builtin_amdgcn_global_load_async_to_lds_b128(
      (__attribute__((address_space(1))) v4i_t*)g,
      (__attribute__((address_space(3))) v4i_t*)l, 0, 0);
#else
  *(uint4*)l = *(const uint4*)g;
#endif
}

static __device__ __forceinline__ void wait_async_lds() {
#if HAVE_ASYNC_LDS
#if __has_builtin(__builtin_amdgcn_s_wait_asynccnt)
  __builtin_amdgcn_s_wait_asynccnt(0);
#else
  asm volatile("s_wait_asynccnt 0x0" ::: "memory");
#endif
#endif
}

static __device__ __forceinline__ __bf16 f2bf(float f) {
  union { float f; uint32_t u; } v; v.f = f;
  uint32_t r = v.u + 0x7FFFu + ((v.u >> 16) & 1u);   // round-to-nearest-even
  union { unsigned short s; __bf16 b; } o; o.s = (unsigned short)(r >> 16);
  return o.b;
}
static __device__ __forceinline__ float bf2f(__bf16 b) { return (float)b; }

union Frag { v16bf v; uint4 q[2]; };

__global__ __launch_bounds__(256)
void cvt_f32_bf16(const float* __restrict__ in, __bf16* __restrict__ out, int n) {
  int i = (blockIdx.x * 256 + threadIdx.x) * 4;
  if (i + 3 < n) {
    const float4 v = *(const float4*)(in + i);
    out[i + 0] = f2bf(v.x); out[i + 1] = f2bf(v.y);
    out[i + 2] = f2bf(v.z); out[i + 3] = f2bf(v.w);
  }
}

// D = act( A1@B1 [+ A2@B2] + bias ) (+ residual terms per mode)
// M=128, N=256, K=256. A: bf16 row-major. B: f32 [256x256] row-major, streamed
// once from HBM, fp32->bf16 converted in registers while staging to LDS.
// NTERMS is compile-time so the double-buffered pipeline fully unrolls:
// constant LDS offsets, immediate global offsets, one barrier per K-step,
// A tile via async global->LDS (ASYNCcnt).
template <int NTERMS>
__global__ __launch_bounds__(256)
void cell_gemm(const __bf16* __restrict__ A1, int lda1,
               const __bf16* __restrict__ A2, int lda2,
               const float* __restrict__ B1, const float* __restrict__ B2,
               const float* __restrict__ bias,
               const __bf16* __restrict__ res0, const __bf16* __restrict__ res1,
               __bf16* __restrict__ outBf,
               float* __restrict__ outF1, long ldF1,
               float* __restrict__ outF2,
               int mode)
{
  __shared__ __attribute__((aligned(16))) __bf16 lsA[2][BM * LDP];
  __shared__ __attribute__((aligned(16))) __bf16 lsB[2][BN * LDP];

  const int tid   = threadIdx.x;
  const int lane  = tid & 31;
  const int wave  = tid >> 5;
  const int lmod  = lane & 15;
  const int lhalf = lane >> 4;
  const int m0 = (blockIdx.x & 1) * BM;    // 2 M-blocks
  const int n0 = (blockIdx.x >> 1) * BN;   // 8 N-blocks
  const int mt = wave & 3;                 // 4 M-tiles per block
  const int nt = wave >> 2;                // 2 N-tiles per block

  v8f acc = {};

  constexpr int NSTEPS = NTERMS * (256 / BK);

  // ---- staging: issue global traffic for K-step s into LDS buffer `buf`
  auto stage = [&](int s, int buf) {
    const int term = s >> 3;
    const int k0   = (s & 7) * BK;
    const __bf16* A = (term == 0) ? A1 : A2;
    const int lda   = (term == 0) ? lda1 : lda2;
    gcfloat_t* Bw   = (gcfloat_t*)((term == 0) ? B1 : B2);
    { // A tile [BM x BK] bf16: async global->LDS, b128 per thread
      const int r = tid >> 2, c = tid & 3;
      async_copy_b128(A + (size_t)(m0 + r) * lda + k0 + c * 8,
                      &lsA[buf][r * LDP + c * 8]);
    }
    { // B tile [BK x BN] f32: coalesced global loads, cvt to bf16, transposed store
      const int n = tid & 31, kc = tid >> 5;    // kc 0..7, 4 K-values each
      __bf16 tmp[4];
#pragma unroll
      for (int i = 0; i < 4; ++i)
        tmp[i] = f2bf(Bw[(size_t)(k0 + kc * 4 + i) * H_ + n0 + n]);
      *(uint2*)&lsB[buf][n * LDP + kc * 4] = *(const uint2*)tmp;
      if (k0 + BK < 256)   // pull the next weight tile toward L2
        __builtin_prefetch((const float*)Bw + (size_t)(k0 + BK) * H_ + n0 + n, 0, 1);
    }
  };

  // ---- compute: one 16x16x32 WMMA from LDS buffer `buf`
  auto compute = [&](int buf) {
    Frag fa, fb;
    // A frag: lane holds row M=mt*16+lmod; K = {kb..kb+7, kb+16..kb+23}, kb=8*lhalf
    const __bf16* pa = &lsA[buf][(mt * 16 + lmod) * LDP + lhalf * 8];
    fa.q[0] = *(const uint4*)pa;
    fa.q[1] = *(const uint4*)(pa + 16);
    // B frag: lane holds col N=nt*16+lmod; K = {kb..kb+7, kb+16..kb+23}, kb=8*lhalf
    const __bf16* pb = &lsB[buf][(nt * 16 + lmod) * LDP + lhalf * 8];
    fb.q[0] = *(const uint4*)pb;
    fb.q[1] = *(const uint4*)(pb + 16);
    acc = __builtin_amdgcn_wmma_f32_16x16x32_bf16(false, fa.v, false, fb.v,
                                                  (short)0, acc, false, false);
  };

  // ---- fully unrolled software pipeline: stage(s+1) overlaps compute(s)
  stage(0, 0);
#pragma unroll
  for (int s = 0; s < NSTEPS; ++s) {
    wait_async_lds();          // our async writes for tile s done
    __syncthreads();           // everyone's tile-s staging visible, tile-(s-1) reads done
    if (s + 1 < NSTEPS) stage(s + 1, (s + 1) & 1);
    compute(s & 1);
  }

  // C/D layout: VGPR i -> M = i + 8*lhalf, N = lmod
  const int col = n0 + nt * 16 + lmod;
  const float bv = ((gcfloat_t*)bias)[col];
#pragma unroll
  for (int i = 0; i < 8; ++i) {
    const int row = m0 + mt * 16 + lhalf * 8 + i;
    const size_t idx = (size_t)row * H_ + col;
    float v = acc[i] + bv;
    float o;
    if (mode == 0) {
      o = tanhf(v);
    } else if (mode == 1) {
      o = fmaxf(v, 0.0f) + bf2f(((gcbf_t*)res0)[idx]);
    } else {
      const float s = 1.0f / (1.0f + expf(-v));
      o = 0.5f * (bf2f(((gcbf_t*)res1)[idx]) + s + bf2f(((gcbf_t*)res0)[idx]));
    }
    ((gbf_t*)outBf)[idx] = f2bf(o);
    if (mode == 2) {
      if (outF1) ((gfloat_t*)outF1)[(size_t)row * ldF1 + col] = o;  // seq out [B,T,H]
      if (outF2) ((gfloat_t*)outF2)[(size_t)row * H_ + col] = o;    // final hidden [B,H]
    }
  }
}

extern "C" void kernel_launch(void* const* d_in, const int* in_sizes, int n_in,
                              void* d_out, int out_size, void* d_ws, size_t ws_size,
                              hipStream_t stream) {
  const float* x      = (const float*)d_in[0];
  const float* hidden = (const float*)d_in[1];
  const float* Win0   = (const float*)d_in[2];
  const float* Wh0    = (const float*)d_in[3];
  const float* b0     = (const float*)d_in[4];
  const float* Win1   = (const float*)d_in[5];
  const float* Wh1    = (const float*)d_in[6];
  const float* b1     = (const float*)d_in[7];
  float* out = (float*)d_out;

  // workspace layout (bf16): x copy, h0, h1, n0, n1  (~4.4 MiB total)
  __bf16* xbf = (__bf16*)d_ws;
  __bf16* h0  = xbf + (size_t)B_ * T_ * D_;
  __bf16* h1  = h0 + (size_t)B_ * H_;
  __bf16* n0b = h1 + (size_t)B_ * H_;
  __bf16* n1b = n0b + (size_t)B_ * H_;

  { int n = B_ * T_ * D_; cvt_f32_bf16<<<n / 1024, 256, 0, stream>>>(x, xbf, n); }
  { int n = 2 * B_ * H_;  cvt_f32_bf16<<<n / 1024, 256, 0, stream>>>(hidden, h0, n); }

  const dim3 grid(16), blk(256);
  float* outSeq = out;                                // [B,T,H]
  float* hid0   = out + (size_t)B_ * T_ * H_;         // hidden[0] [B,H]
  float* hid1   = hid0 + (size_t)B_ * H_;             // hidden[1] [B,H]

  for (int t = 0; t < T_; ++t) {
    // ---------- layer 0 ----------
    const __bf16* xt = xbf + (size_t)t * D_;          // row stride T*D
    const float* W0i = Win0 + (size_t)t * D_ * H_;
    const float* W0h = Wh0 + (size_t)t * 3 * H_ * H_;
    const float* bb0 = b0 + (size_t)t * 3 * H_;
    cell_gemm<2><<<grid, blk, 0, stream>>>(xt, T_ * D_, h0, H_, W0i, W0h,
        bb0, nullptr, nullptr, n0b, nullptr, 0, nullptr, 0);
    cell_gemm<1><<<grid, blk, 0, stream>>>(n0b, H_, nullptr, 0,
        W0h + (size_t)H_ * H_, nullptr, bb0 + H_,
        n0b, nullptr, n1b, nullptr, 0, nullptr, 1);
    cell_gemm<1><<<grid, blk, 0, stream>>>(n1b, H_, nullptr, 0,
        W0h + (size_t)2 * H_ * H_, nullptr, bb0 + 2 * H_,
        n0b, n1b, h0, nullptr, 0, (t == T_ - 1) ? hid0 : nullptr, 2);
    // ---------- layer 1 (input = h0 just written = out0[t]) ----------
    const float* W1i = Win1 + (size_t)t * H_ * H_;
    const float* W1h = Wh1 + (size_t)t * 3 * H_ * H_;
    const float* bb1 = b1 + (size_t)t * 3 * H_;
    cell_gemm<2><<<grid, blk, 0, stream>>>(h0, H_, h1, H_, W1i, W1h,
        bb1, nullptr, nullptr, n0b, nullptr, 0, nullptr, 0);
    cell_gemm<1><<<grid, blk, 0, stream>>>(n0b, H_, nullptr, 0,
        W1h + (size_t)H_ * H_, nullptr, bb1 + H_,
        n0b, nullptr, n1b, nullptr, 0, nullptr, 1);
    cell_gemm<1><<<grid, blk, 0, stream>>>(n1b, H_, nullptr, 0,
        W1h + (size_t)2 * H_ * H_, nullptr, bb1 + 2 * H_,
        n0b, n1b, h1, outSeq + (size_t)t * H_, (long)T_ * H_,
        (t == T_ - 1) ? hid1 : nullptr, 2);
  }
  (void)in_sizes; (void)n_in; (void)out_size; (void)ws_size;
}
